// MultiheadAttention_53300544143913
// MI455X (gfx1250) — compile-verified, matrix operand order checked by reading, not run on hardware
//
#include <hip/hip_runtime.h>

// ---------------------------------------------------------------------------
// MultiheadAttention for MI455X (gfx1250, wave32, WMMA bf16 + TDM path)
// B=4, N=2048, C=512, H=8, HD=64.
//  - All GEMMs + attention via v_wmma_f32_16x16x32_bf16 (f32 accumulation)
//  - Flash attention (online softmax): 2048x2048 scores never hit HBM
//  - K/V tiles double-buffered in LDS via tensor_load_to_lds (TDM);
//    V projected directly in transposed [B,H,HD,N] layout so both TDM tiles
//    are plain row-major 2-D tiles.
//  - GEMM k-step loads A + all 4 B fragments up-front so the scheduler can
//    overlap loads with WMMAs using partial s_wait_loadcnt waits.
// ---------------------------------------------------------------------------

typedef __bf16 bf16;
typedef __attribute__((ext_vector_type(16))) __bf16 v16bf;
typedef __attribute__((ext_vector_type(8)))  __bf16 v8bf;
typedef __attribute__((ext_vector_type(8)))  float  v8f;
typedef __attribute__((ext_vector_type(4)))  unsigned int u32x4;
typedef __attribute__((ext_vector_type(4)))  int i32x4;
typedef __attribute__((ext_vector_type(8)))  int i32x8;

#define B_   4
#define N_   2048
#define C_   512
#define H_   8
#define HD_  64
#define SCALE_ 0.125f      // HD^-0.5
#define NEGINF_ (-1e30f)

// ---- TDM availability -----------------------------------------------------
#ifdef __has_builtin
#if __has_builtin(__builtin_amdgcn_tensor_load_to_lds) && \
    __has_builtin(__builtin_amdgcn_s_wait_tensorcnt)
#define HAVE_TDM 1
#endif
#endif
#ifndef HAVE_TDM
#define HAVE_TDM 0
#endif

#if HAVE_TDM
#if __clang_major__ >= 23
// amdgpu-toolchain / clang-23: 6-arg form (extra i32x8 group)
#define TDM_LOAD(g0, g1)                                                      \
  __builtin_amdgcn_tensor_load_to_lds((g0), (g1), (i32x4){0, 0, 0, 0},        \
                                      (i32x4){0, 0, 0, 0},                    \
                                      (i32x8){0, 0, 0, 0, 0, 0, 0, 0}, 0)
#else
// ROCm 7.2 / clang-22: 5-arg form
#define TDM_LOAD(g0, g1)                                                      \
  __builtin_amdgcn_tensor_load_to_lds((g0), (g1), (i32x4){0, 0, 0, 0},        \
                                      (i32x4){0, 0, 0, 0}, 0)
#endif

// D# group 0: count=1 (user), lds_addr, 57-bit global addr, type=2 ("image")
static __device__ __forceinline__ u32x4 tdm_g0(unsigned lds_off,
                                               const void* gptr) {
  unsigned long long ga = (unsigned long long)(size_t)gptr;
  u32x4 g;
  g[0] = 1u;
  g[1] = lds_off;
  g[2] = (unsigned)ga;
  g[3] = (unsigned)((ga >> 32) & 0x1FFFFFFu) | (2u << 30);
  return g;
}

// D# group 1 for a 2-D tile of 2-byte elements (ISA 8.4 bit packing)
static __device__ __forceinline__ i32x8 tdm_g1(int tdim0, int tdim1,
                                               int tile0, int tile1,
                                               int stride0) {
  i32x8 g = {0, 0, 0, 0, 0, 0, 0, 0};
  g[0] = 1 << 16;                                     // data_size = 2 bytes
  g[1] = (tdim0 & 0xFFFF) << 16;                      // tensor_dim0[15:0]
  g[2] = ((tdim0 >> 16) & 0xFFFF) | ((tdim1 & 0xFFFF) << 16);
  g[3] = ((tdim1 >> 16) & 0xFFFF) | ((tile0 & 0xFFFF) << 16);
  g[4] = (tile1 & 0xFFFF);                            // tile_dim1 (tile_dim2=0)
  g[5] = stride0;                                     // tensor_dim0_stride lo
  return g;
}
#endif  // HAVE_TDM

// ---------------------------------------------------------------------------
// Fragment loader: 16x32 K-striped WMMA fragment (ISA 7.12.2, 16-bit A 16x32).
// Lane L (<16): row M=L, K={0..7,16..23}; lane L+16: row M=L, K={8..15,24..31}.
// Also serves as the B-matrix loader when column n of B is a contiguous
// 32-element run at base + n*ld.
// ---------------------------------------------------------------------------
template <typename P>
static __device__ __forceinline__ v16bf frag_ld(P base, int ld) {
  const int lane = threadIdx.x & 31;
  const int m    = lane & 15;
  const int kb   = (lane >> 4) << 3;     // 0 or 8
  P p = base + m * ld + kb;
  v8bf lo = *(const v8bf*)(p);
  v8bf hi = *(const v8bf*)(p + 16);
  v16bf f;
#pragma unroll
  for (int i = 0; i < 8; ++i) { f[i] = lo[i]; f[i + 8] = hi[i]; }
  return f;
}

static __device__ __forceinline__ v8f wmma_bf16(v16bf a, v16bf b, v8f c) {
  return __builtin_amdgcn_wmma_f32_16x16x32_bf16(false, a, false, b,
                                                 (short)0, c, false, false);
}

// ---------------------------------------------------------------------------
// fp32 -> bf16 conversion
// ---------------------------------------------------------------------------
__global__ void cvt_f32_bf16(const float* __restrict__ s,
                             bf16* __restrict__ d, int n) {
  int i = (blockIdx.x * blockDim.x + threadIdx.x) * 4;
  if (i + 3 < n) {
    float4 f = *(const float4*)(s + i);
    d[i + 0] = (bf16)f.x; d[i + 1] = (bf16)f.y;
    d[i + 2] = (bf16)f.z; d[i + 3] = (bf16)f.w;
  }
}

// ---------------------------------------------------------------------------
// QKV projection: each wave owns a 16x64 strip (A fragment reused 4x).
// grid.x = (M/64)*(C/64), grid.y picks q/k/v.  q,k -> [B,H,N,HD];
// v -> transposed [B,H,HD,N].
// ---------------------------------------------------------------------------
__global__ __launch_bounds__(128)
void proj_kernel(const bf16* __restrict__ x0, const bf16* __restrict__ x1,
                 const bf16* __restrict__ x2,
                 const bf16* __restrict__ w0, const bf16* __restrict__ w1,
                 const bf16* __restrict__ w2,
                 bf16* __restrict__ d0, bf16* __restrict__ d1,
                 bf16* __restrict__ d2) {
  const bf16* X = (blockIdx.y == 0) ? x0 : (blockIdx.y == 1) ? x1 : x2;
  const bf16* W = (blockIdx.y == 0) ? w0 : (blockIdx.y == 1) ? w1 : w2;
  bf16*       D = (blockIdx.y == 0) ? d0 : (blockIdx.y == 1) ? d1 : d2;

  const int wave = threadIdx.x >> 5;
  const int lane = threadIdx.x & 31;
  const int mt   = blockIdx.x >> 3;       // 128 row blocks of 64
  const int cb   = blockIdx.x & 7;        // 8 col blocks of 64
  const int row0 = mt * 64 + wave * 16;
  const int col0 = cb * 64;

  v8f acc0 = {}, acc1 = {}, acc2 = {}, acc3 = {};
#pragma unroll 2
  for (int k = 0; k < C_; k += 32) {
    // distinct live fragments -> loads stay in flight, partial loadcnt waits
    v16bf a  = frag_ld(X + row0 * C_ + k, C_);
    v16bf b0 = frag_ld(W + (col0 +  0) * C_ + k, C_);
    v16bf b1 = frag_ld(W + (col0 + 16) * C_ + k, C_);
    v16bf b2 = frag_ld(W + (col0 + 32) * C_ + k, C_);
    v16bf b3 = frag_ld(W + (col0 + 48) * C_ + k, C_);
    acc0 = wmma_bf16(a, b0, acc0);
    acc1 = wmma_bf16(a, b1, acc1);
    acc2 = wmma_bf16(a, b2, acc2);
    acc3 = wmma_bf16(a, b3, acc3);
  }

  const int hL = lane >> 4, nc = lane & 15;
  const int h  = cb;                      // 64-wide col block == head
  v8f acc[4] = {acc0, acc1, acc2, acc3};
#pragma unroll
  for (int j = 0; j < 4; ++j) {
    const int dd = 16 * j + nc;
#pragma unroll
    for (int r = 0; r < 8; ++r) {
      int row = row0 + r + 8 * hL;
      int bb  = row >> 11, n = row & (N_ - 1);
      if (blockIdx.y == 2)                 // V: [B,H,HD,N]
        D[(((size_t)(bb * H_ + h) * HD_ + dd) * N_) + n] = (bf16)acc[j][r];
      else                                 // Q,K: [B,H,N,HD]
        D[(((size_t)(bb * H_ + h) * N_ + n) * HD_) + dd] = (bf16)acc[j][r];
    }
  }
}

// ---------------------------------------------------------------------------
// Flash attention.  grid = (N/64, H, B); 4 waves x 16 q-rows.
// K tile [32x64] and V^T tile [64x32] double-buffered in LDS via TDM
// (fallback: direct vector copies).  Per 32-key tile: 4 WMMAs QK^T,
// online softmax, 4 WMMAs PV.
// ---------------------------------------------------------------------------
__global__ __launch_bounds__(128)
void flash_kernel(const bf16* __restrict__ q, const bf16* __restrict__ k,
                  const bf16* __restrict__ vt,
                  const float* __restrict__ amask,
                  const unsigned char* __restrict__ pad,   // jax bool = 1B
                  bf16* __restrict__ xo) {
  const int b = blockIdx.z, h = blockIdx.y;
  const int wave = threadIdx.x >> 5, lane = threadIdx.x & 31;
  const int hL = lane >> 4, nc = lane & 15;
  const int q0 = blockIdx.x * 64 + wave * 16;

  const bf16* qb  = q  + (size_t)((b * H_ + h) * N_) * HD_;
  const bf16* kb  = k  + (size_t)((b * H_ + h) * N_) * HD_;
  const bf16* vtb = vt + (size_t)((b * H_ + h) * HD_) * N_;  // [HD][N]

  __shared__ __align__(16) bf16 Ks[2][32][64];   // keys x dims
  __shared__ __align__(16) bf16 Vt[2][64][32];   // dims x keys
  __shared__ __align__(16) bf16 Ps[4][16][32];   // per-wave P tile

  v16bf aQ0 = frag_ld(qb + q0 * HD_ + 0,  HD_);
  v16bf aQ1 = frag_ld(qb + q0 * HD_ + 32, HD_);

  v8f o0 = {}, o1 = {}, o2 = {}, o3 = {};
  float m_[8], l_[8];
#pragma unroll
  for (int r = 0; r < 8; ++r) { m_[r] = NEGINF_; l_[r] = 0.f; }

#if HAVE_TDM
  const i32x8 g1K = tdm_g1(HD_, N_, HD_, 32, HD_);   // 32 rows x 64 dims
  const i32x8 g1V = tdm_g1(N_, HD_, 32, HD_, N_);    // 64 dims x 32 keys
  if (wave == 0) {   // prologue: tile 0 -> buffer 0
    TDM_LOAD(tdm_g0((unsigned)(size_t)&Ks[0][0][0], kb), g1K);
    TDM_LOAD(tdm_g0((unsigned)(size_t)&Vt[0][0][0], vtb), g1V);
  }
#endif
  int cur = 0;

  for (int kc = 0; kc < N_; kc += 32) {
#if HAVE_TDM
    if (wave == 0) {
      const int kn = (kc + 32) & (N_ - 1);   // wrap: tail issue is harmless
      TDM_LOAD(tdm_g0((unsigned)(size_t)&Ks[cur ^ 1][0][0],
                      kb + (size_t)kn * HD_), g1K);
      TDM_LOAD(tdm_g0((unsigned)(size_t)&Vt[cur ^ 1][0][0], vtb + kn), g1V);
      __builtin_amdgcn_s_wait_tensorcnt(2);  // current tile has landed
    }
    __syncthreads();
#else
    {   // cooperative copy: 128 threads x 16 bf16 each per tile
      const int e0 = threadIdx.x * 16;
      const int rk = e0 >> 6, dk = e0 & 63;          // K tile: 32x64
      *(v8bf*)&Ks[0][rk][dk]     = *(const v8bf*)(kb + (kc + rk) * HD_ + dk);
      *(v8bf*)&Ks[0][rk][dk + 8] = *(const v8bf*)(kb + (kc + rk) * HD_ + dk + 8);
      const int dv = e0 >> 5, cv = e0 & 31;          // V^T tile: 64x32
      *(v8bf*)&Vt[0][dv][cv]     = *(const v8bf*)(vtb + (size_t)dv * N_ + kc + cv);
      *(v8bf*)&Vt[0][dv][cv + 8] = *(const v8bf*)(vtb + (size_t)dv * N_ + kc + cv + 8);
    }
    __syncthreads();
#endif

    // ---- S = Q @ K^T : two 16x16 C tiles over head-dim 64 ----
    v8f s0 = {}, s1 = {};
    s0 = wmma_bf16(aQ0, frag_ld(&Ks[cur][0][0],  64), s0);
    s0 = wmma_bf16(aQ1, frag_ld(&Ks[cur][0][32], 64), s0);
    s1 = wmma_bf16(aQ0, frag_ld(&Ks[cur][16][0],  64), s1);
    s1 = wmma_bf16(aQ1, frag_ld(&Ks[cur][16][32], 64), s1);

    // ---- mask + online softmax ----
    const int c0 = kc + nc;
    const float pm0 = pad[b * N_ + c0]      ? NEGINF_ : 0.f;
    const float pm1 = pad[b * N_ + c0 + 16] ? NEGINF_ : 0.f;
    float corr[8];
#pragma unroll
    for (int r = 0; r < 8; ++r) {
      const int row = q0 + r + 8 * hL;
      float v0 = s0[r] * SCALE_ + amask[(size_t)row * N_ + c0]      + pm0;
      float v1 = s1[r] * SCALE_ + amask[(size_t)row * N_ + c0 + 16] + pm1;
      float t = fmaxf(v0, v1);
#pragma unroll
      for (int sft = 1; sft <= 8; sft <<= 1)
        t = fmaxf(t, __shfl_xor(t, sft, 32));      // within 16-lane half
      const float mn = fmaxf(m_[r], t);
      const float cr = __expf(m_[r] - mn);
      const float p0 = __expf(v0 - mn);
      const float p1 = __expf(v1 - mn);
      float rs = p0 + p1;
#pragma unroll
      for (int sft = 1; sft <= 8; sft <<= 1)
        rs += __shfl_xor(rs, sft, 32);
      l_[r] = l_[r] * cr + rs;
      m_[r] = mn;
      corr[r] = cr;
      Ps[wave][r + 8 * hL][nc]      = (bf16)p0;    // C-layout -> A-layout
      Ps[wave][r + 8 * hL][nc + 16] = (bf16)p1;    //   via per-wave LDS tile
    }
#pragma unroll
    for (int r = 0; r < 8; ++r) {
      o0[r] *= corr[r]; o1[r] *= corr[r];
      o2[r] *= corr[r]; o3[r] *= corr[r];
    }

    asm volatile("s_wait_dscnt 0" ::: "memory");   // wave-private round-trip
    v16bf aP = frag_ld(&Ps[wave][0][0], 32);

    // ---- O += P @ V ----
    o0 = wmma_bf16(aP, frag_ld(&Vt[cur][0][0],  32), o0);
    o1 = wmma_bf16(aP, frag_ld(&Vt[cur][16][0], 32), o1);
    o2 = wmma_bf16(aP, frag_ld(&Vt[cur][32][0], 32), o2);
    o3 = wmma_bf16(aP, frag_ld(&Vt[cur][48][0], 32), o3);
    __syncthreads();   // all waves done with `cur` before TDM overwrites it
#if HAVE_TDM
    cur ^= 1;
#endif
  }

  // ---- normalize + write [B,N,C] bf16 ----
#pragma unroll
  for (int r = 0; r < 8; ++r) {
    const float inv = 1.f / fmaxf(l_[r], 1e-30f);
    const int row = q0 + r + 8 * hL;
    bf16* dst = xo + (size_t)(b * N_ + row) * C_ + h * HD_ + nc;
    dst[0]  = (bf16)(o0[r] * inv);
    dst[16] = (bf16)(o1[r] * inv);
    dst[32] = (bf16)(o2[r] * inv);
    dst[48] = (bf16)(o3[r] * inv);
  }
}

// ---------------------------------------------------------------------------
// Output projection: out = xo @ Wo.T + bo (fp32), 16x64 strip per wave
// ---------------------------------------------------------------------------
__global__ __launch_bounds__(128)
void oproj_kernel(const bf16* __restrict__ X, const bf16* __restrict__ W,
                  const float* __restrict__ bo, float* __restrict__ out) {
  const int wave = threadIdx.x >> 5;
  const int lane = threadIdx.x & 31;
  const int mt   = blockIdx.x >> 3;
  const int cb   = blockIdx.x & 7;
  const int row0 = mt * 64 + wave * 16;
  const int col0 = cb * 64;

  v8f acc0 = {}, acc1 = {}, acc2 = {}, acc3 = {};
#pragma unroll 2
  for (int k = 0; k < C_; k += 32) {
    v16bf a  = frag_ld(X + row0 * C_ + k, C_);
    v16bf b0 = frag_ld(W + (col0 +  0) * C_ + k, C_);
    v16bf b1 = frag_ld(W + (col0 + 16) * C_ + k, C_);
    v16bf b2 = frag_ld(W + (col0 + 32) * C_ + k, C_);
    v16bf b3 = frag_ld(W + (col0 + 48) * C_ + k, C_);
    acc0 = wmma_bf16(a, b0, acc0);
    acc1 = wmma_bf16(a, b1, acc1);
    acc2 = wmma_bf16(a, b2, acc2);
    acc3 = wmma_bf16(a, b3, acc3);
  }

  const int hL = lane >> 4, nc = lane & 15;
  v8f acc[4] = {acc0, acc1, acc2, acc3};
#pragma unroll
  for (int j = 0; j < 4; ++j) {
    const int col = col0 + 16 * j + nc;
    const float bias = bo[col];
#pragma unroll
    for (int r = 0; r < 8; ++r) {
      int row = row0 + r + 8 * hL;
      out[(size_t)row * C_ + col] = acc[j][r] + bias;
    }
  }
}

// ---------------------------------------------------------------------------
// launch
// ---------------------------------------------------------------------------
extern "C" void kernel_launch(void* const* d_in, const int* in_sizes, int n_in,
                              void* d_out, int out_size, void* d_ws, size_t ws_size,
                              hipStream_t stream) {
  const float* query = (const float*)d_in[0];
  const float* key_  = (const float*)d_in[1];
  const float* value = (const float*)d_in[2];
  const float* amask = (const float*)d_in[3];
  const unsigned char* pad = (const unsigned char*)d_in[4];  // jax bool
  const float* Wq = (const float*)d_in[5];
  const float* Wk = (const float*)d_in[6];
  const float* Wv = (const float*)d_in[7];
  const float* Wo = (const float*)d_in[8];
  const float* bo = (const float*)d_in[9];
  float* out = (float*)d_out;

  const int NTOK = B_ * N_ * C_;   // 4,194,304
  const int NW   = C_ * C_;        // 262,144

  bf16* xq = (bf16*)d_ws;
  bf16* xk = xq + NTOK;
  bf16* xv = xk + NTOK;
  bf16* wq = xv + NTOK;
  bf16* wk = wq + NW;
  bf16* wv = wk + NW;
  bf16* wo = wv + NW;
  bf16* qp = wo + NW;              // [B,H,N,HD]
  bf16* kp = qp + NTOK;
  bf16* vp = kp + NTOK;            // V transposed: [B,H,HD,N]
  bf16* xo = vp + NTOK;            // [B,N,C] attention output (bf16)

  // 1) fp32 -> bf16 copies
  cvt_f32_bf16<<<NTOK / 1024, 256, 0, stream>>>(query, xq, NTOK);
  cvt_f32_bf16<<<NTOK / 1024, 256, 0, stream>>>(key_,  xk, NTOK);
  cvt_f32_bf16<<<NTOK / 1024, 256, 0, stream>>>(value, xv, NTOK);
  cvt_f32_bf16<<<NW / 1024, 256, 0, stream>>>(Wq, wq, NW);
  cvt_f32_bf16<<<NW / 1024, 256, 0, stream>>>(Wk, wk, NW);
  cvt_f32_bf16<<<NW / 1024, 256, 0, stream>>>(Wv, wv, NW);
  cvt_f32_bf16<<<NW / 1024, 256, 0, stream>>>(Wo, wo, NW);

  // 2) Q/K/V projections (V written transposed)
  dim3 pg((B_ * N_ / 64) * (C_ / 64), 3);
  proj_kernel<<<pg, 128, 0, stream>>>(xq, xk, xv, wq, wk, wv, qp, kp, vp);

  // 3) flash attention (TDM double-buffered K/V^T tiles)
  dim3 fg(N_ / 64, H_, B_);
  flash_kernel<<<fg, 128, 0, stream>>>(qp, kp, vp, amask, pad, xo);

  // 4) output projection + bias (fp32 out)
  oproj_kernel<<<(B_ * N_ / 64) * (C_ / 64), 128, 0, stream>>>(xo, wo, bo, out);
}
